// Model_71339406787341
// MI455X (gfx1250) — matrix operand order repaired and checked by reading, not compiled, not run on hardware
//
#include <hip/hip_runtime.h>

// CDNA5 / gfx1250: wave32, WMMA bf16 16x16x32 path.
typedef __attribute__((ext_vector_type(16))) __bf16 v16bf;
typedef __attribute__((ext_vector_type(8)))  float  v8f;
typedef __attribute__((ext_vector_type(4)))  int    v4i;

typedef __attribute__((address_space(1))) v4i gv4i;   // global 16B vector
typedef __attribute__((address_space(3))) v4i lv4i;   // LDS 16B vector

// Detect the gfx1250 async global->LDS copy builtin (ASYNCcnt path).
#define USE_ASYNC_LDS 0
#if defined(__has_builtin)
#if __has_builtin(__builtin_amdgcn_global_load_async_to_lds_b128)
#undef USE_ASYNC_LDS
#define USE_ASYNC_LDS 1
#endif
#endif

// ---------------------------------------------------------------------------
// Weight pre-pack: OIHW fp32 -> bf16 packed per WMMA B-matrix (32x16) layout.
// Packed index: [ntile][tap r][chunk ck][lane][e]; for lane L (N = L%16,
// half = L/16) element e corresponds to K = (e/8)*16 + half*8 + (e%8) (16-bit B layout:
// lanes0-15 hold K=0..7,16..23; lanes16-31 hold K=8..15,24..31).
// Channel = ck*32 + K; zero-padded beyond CIN/COUT.
// ---------------------------------------------------------------------------
template<int CIN, int CHUNKS, int COUT>
__global__ __launch_bounds__(256) void pack_w(const float* __restrict__ w,
                                              __bf16* __restrict__ wp) {
    constexpr int NT = (COUT + 15) / 16;
    const int t = blockIdx.x * 256 + threadIdx.x;
    const int total = NT * 9 * CHUNKS * 32;
    if (t >= total) return;
    const int lane = t & 31;
    const int ck   = (t >> 5) % CHUNKS;
    const int r    = (t / (32 * CHUNKS)) % 9;
    const int nt   = t / (32 * CHUNKS * 9);
    const int kh = r / 3, kw = r % 3;
    const int o    = nt * 16 + (lane & 15);
    const int half = lane >> 4;

    __bf16* dst = wp + (size_t)nt * (9 * CHUNKS * 512)
                     + ((size_t)(r * CHUNKS + ck) * 32 + lane) * 16;
#pragma unroll
    for (int e = 0; e < 16; ++e) {
        const int c = ck * 32 + (e >> 3) * 16 + half * 8 + (e & 7);
        float v = 0.0f;
        if (o < COUT && c < CIN)
            v = w[(((size_t)o * CIN + c) * 3 + kh) * 3 + kw];
        dst[e] = (__bf16)v;
    }
}

// Stage nelem bf16 (multiple of 8) from global to LDS, whole block.
__device__ __forceinline__ void stage_to_lds(const __bf16* __restrict__ g,
                                             __bf16* s, int nelem) {
    const int tid  = threadIdx.x;
    const int nvec = nelem / 8;                         // 16B units
#if USE_ASYNC_LDS
    for (int i = tid; i < nvec; i += 256) {
        __builtin_amdgcn_global_load_async_to_lds_b128(
            (gv4i*)(g + (size_t)i * 8),
            (lv4i*)(s + (size_t)i * 8),
            0, 0);
    }
#if __has_builtin(__builtin_amdgcn_s_wait_asynccnt)
    __builtin_amdgcn_s_wait_asynccnt(0);
#else
    asm volatile("s_wait_asynccnt 0" ::: "memory");
#endif
#else
    for (int i = tid; i < nvec; i += 256)
        ((v4i*)s)[i] = ((const v4i*)g)[i];
#endif
    __syncthreads();
}

// ---------------------------------------------------------------------------
// Implicit-GEMM conv. One wave computes a 16(Mpix) x (NTB*16)(Nout) strip:
// A fetched once per (tap,chunk) and reused across NTB WMMAs whose B tiles
// live in LDS. 8 waves/block (8 consecutive M tiles).
//   IN_MODE  0: bf16 NHWC input (nearest-up2 fused via UP)
//            1: fp32 NCHW input (first layer)
//   OUT_MODE 0: ReLU -> bf16 NHWC
//            1: sigmoid -> fp32 NCHW (final layer)
// A-operand per lane (16-bit A 16x32 layout): row M = lane%16,
//   element e -> K = (e/8)*16 + (lane/16)*8 + (e%8)  => two contiguous
//   8-channel (16B) NHWC loads per 32-wide K chunk.
// ---------------------------------------------------------------------------
template<int CIN, int CHUNKS, int COUT, int STRIDE, int UP, int IN_MODE,
         int OUT_MODE, int NTB, int WOUT>
__global__ __launch_bounds__(256) void conv_wmma(const void* __restrict__ in,
                                                 const __bf16* __restrict__ wpack,
                                                 const float* __restrict__ bias,
                                                 void* __restrict__ out,
                                                 int Hsrc, int Wsrc, int Hout) {
    constexpr int TILE_ELEMS = 9 * CHUNKS * 512;        // bf16 per N-tile
    constexpr int WELEMS = NTB * TILE_ELEMS;            // <= 36 KB
    __shared__ __bf16 sW[WELEMS];

    const int tid    = threadIdx.x;
    const int lane   = tid & 31;
    const int wave   = tid >> 5;
    const int half   = lane >> 4;
    const int lane16 = lane & 15;
    const int ntbase = blockIdx.y * NTB;
    const int b      = blockIdx.z;
    const int mbase  = (blockIdx.x * 8 + wave) * 16;

    // Stage NTB N-tiles of packed weights into LDS (shared by all 8 waves).
    const __bf16* wg = wpack + (size_t)ntbase * TILE_ELEMS;
    __builtin_prefetch(wg, 0, 0);                       // global_prefetch_b8
    stage_to_lds(wg, sW, WELEMS);

    // This lane's A-row output pixel.
    const int pm  = mbase + lane16;
    const int ohA = pm / WOUT, owA = pm % WOUT;
    const int Hin = Hsrc * UP, Win = Wsrc * UP;

    // Accumulators = bias broadcast (C/D layout: lane N = lane%16).
    v8f acc[NTB];
#pragma unroll
    for (int nb = 0; nb < NTB; ++nb) {
        const int ncol = (ntbase + nb) * 16 + lane16;
        const float bn = (ncol < COUT) ? bias[ncol] : 0.0f;
#pragma unroll
        for (int e = 0; e < 8; ++e) acc[nb][e] = bn;
    }

#pragma unroll
    for (int kh = 0; kh < 3; ++kh) {
        const int ih = ohA * STRIDE + kh - 1;
        const bool rowok = (ih >= 0) && (ih < Hin);
        const int sh = rowok ? ih / UP : 0;
#pragma unroll
        for (int kw = 0; kw < 3; ++kw) {
            const int iw = owA * STRIDE + kw - 1;
            const bool ok = rowok && (iw >= 0) && (iw < Win);
            const int sw = ok ? iw / UP : 0;
            const int r  = kh * 3 + kw;
#pragma unroll
            for (int ck = 0; ck < CHUNKS; ++ck) {
                v16bf a;
                if (IN_MODE == 1) {
                    // fp32 NCHW gather (CIN=3): most elements zero-padded.
                    const float* xin = (const float*)in;
#pragma unroll
                    for (int e = 0; e < 16; ++e) {
                        const int c = ck * 32 + (e >> 3) * 16 + half * 8 + (e & 7);
                        float v = 0.0f;
                        if (ok && c < CIN)
                            v = xin[(((size_t)b * CIN + c) * Hsrc + sh) * Wsrc + sw];
                        a[e] = (__bf16)v;
                    }
                } else {
                    // bf16 NHWC: two 16B channel-group loads per lane.
                    const __bf16* base = (const __bf16*)in
                        + (((size_t)b * Hsrc + sh) * Wsrc + sw) * CIN;
                    const int c0 = ck * 32 + half * 8;       // elems 0..7
                    const int c1 = c0 + 16;                  // elems 8..15
                    v4i z = {0, 0, 0, 0};
                    v4i g0 = (ok && (c0 + 8 <= CIN)) ? *(const v4i*)(base + c0) : z;
                    v4i g1 = (ok && (c1 + 8 <= CIN)) ? *(const v4i*)(base + c1) : z;
                    ((v4i*)&a)[0] = g0;
                    ((v4i*)&a)[1] = g1;
                }
                // Reuse A across NTB N-tiles; B is a contiguous 32B LDS read.
#pragma unroll
                for (int nb = 0; nb < NTB; ++nb) {
                    const v16bf bb = *(const v16bf*)(sW + (size_t)nb * TILE_ELEMS
                                     + ((size_t)(r * CHUNKS + ck) * 32 + lane) * 16);
                    acc[nb] = __builtin_amdgcn_wmma_f32_16x16x32_bf16(
                                  false, a, false, bb, (short)0, acc[nb], false, false);
                }
            }
        }
    }

    // Epilogue. C/D layout: VGPR rr holds M = rr + 8*half, N = lane%16.
#pragma unroll
    for (int nb = 0; nb < NTB; ++nb) {
        const int ncol = (ntbase + nb) * 16 + lane16;
#pragma unroll
        for (int rr = 0; rr < 8; ++rr) {
            const int m  = mbase + rr + half * 8;
            const int oh = m / WOUT, ow = m % WOUT;
            float v = acc[nb][rr];
            if (OUT_MODE == 0) {
                if (ncol < COUT) {
                    v = fmaxf(v, 0.0f);
                    ((__bf16*)out)[(((size_t)b * Hout + oh) * WOUT + ow) * COUT + ncol]
                        = (__bf16)v;
                }
            } else {
                if (ncol < COUT) {
                    v = 1.0f / (1.0f + __expf(-v));
                    ((float*)out)[(((size_t)b * COUT + ncol) * Hout + oh) * WOUT + ow] = v;
                }
            }
        }
    }
}

// ---------------------------------------------------------------------------
// Launch: x -> conv1(s2,relu) -> conv2(s2,relu) -> [up2 fused] conv3(relu)
//           -> [up2 fused] conv4 -> sigmoid.  Intermediates: bf16 NHWC in ws.
// ---------------------------------------------------------------------------
extern "C" void kernel_launch(void* const* d_in, const int* in_sizes, int n_in,
                              void* d_out, int out_size, void* d_ws, size_t ws_size,
                              hipStream_t stream) {
    (void)in_sizes; (void)n_in; (void)out_size; (void)ws_size;
    const float* x  = (const float*)d_in[0];
    const float* w1 = (const float*)d_in[1]; const float* b1 = (const float*)d_in[2];
    const float* w2 = (const float*)d_in[3]; const float* b2 = (const float*)d_in[4];
    const float* w3 = (const float*)d_in[5]; const float* b3 = (const float*)d_in[6];
    const float* w4 = (const float*)d_in[7]; const float* b4 = (const float*)d_in[8];

    char* ws = (char*)d_ws;
    size_t off = 0;
    __bf16* h1 = (__bf16*)(ws + off); off += (size_t)64 * 128 * 128 * 48 * 2; // NHWC
    __bf16* h2 = (__bf16*)(ws + off); off += (size_t)64 * 64  * 64  * 64 * 2; // NHWC
    __bf16* h3 = (__bf16*)(ws + off); off += (size_t)64 * 128 * 128 * 32 * 2; // NHWC
    __bf16* wp1 = (__bf16*)(ws + off); off += (size_t)3 * 9 * 1 * 512 * 2;
    __bf16* wp2 = (__bf16*)(ws + off); off += (size_t)4 * 9 * 2 * 512 * 2;
    __bf16* wp3 = (__bf16*)(ws + off); off += (size_t)2 * 9 * 2 * 512 * 2;
    __bf16* wp4 = (__bf16*)(ws + off); off += (size_t)1 * 9 * 1 * 512 * 2;

    // Pack weights into WMMA B layout (tiny kernels).
    pack_w<3, 1, 48><<<dim3(4), dim3(256), 0, stream>>>(w1, wp1);
    pack_w<48, 2, 64><<<dim3(9), dim3(256), 0, stream>>>(w2, wp2);
    pack_w<64, 2, 32><<<dim3(5), dim3(256), 0, stream>>>(w3, wp3);
    pack_w<32, 1, 3><<<dim3(2), dim3(256), 0, stream>>>(w4, wp4);

    // conv1: fp32 NCHW [64,3,256,256] -> bf16 NHWC [64,128,128,48], ReLU.
    //        NTB=3: all 3 N-tiles per wave, A gathered once.
    conv_wmma<3, 1, 48, 2, 1, 1, 0, 3, 128>
        <<<dim3(128, 1, 64), dim3(256), 0, stream>>>(x, wp1, b1, h1, 256, 256, 128);
    // conv2: -> bf16 NHWC [64,64,64,64], ReLU. NTB=2 (2 groups of 2 N-tiles).
    conv_wmma<48, 2, 64, 2, 1, 0, 0, 2, 64>
        <<<dim3(32, 2, 64), dim3(256), 0, stream>>>(h1, wp2, b2, h2, 128, 128, 64);
    // conv3 with fused up2 input: -> bf16 NHWC [64,128,128,32], ReLU. NTB=2.
    conv_wmma<64, 2, 32, 1, 2, 0, 0, 2, 128>
        <<<dim3(128, 1, 64), dim3(256), 0, stream>>>(h2, wp3, b3, h3, 64, 64, 128);
    // conv4 with fused up2 input: -> fp32 NCHW [64,3,256,256], sigmoid. NTB=1.
    conv_wmma<32, 1, 3, 1, 2, 0, 1, 1, 256>
        <<<dim3(512, 1, 64), dim3(256), 0, stream>>>(h3, wp4, b4, d_out, 128, 128, 256);
}
